// LISTA_CPSS_34600256536591
// MI455X (gfx1250) — matrix operand order repaired and checked by reading, not compiled, not run on hardware
//
#include <hip/hip_runtime.h>
#include <cstdint>

typedef __attribute__((ext_vector_type(16))) __bf16 v16bf;
typedef __attribute__((ext_vector_type(8)))  __bf16 v8bf;
typedef __attribute__((ext_vector_type(4)))  __bf16 v4bf;
typedef __attribute__((ext_vector_type(8)))  float  v8f;
typedef __attribute__((ext_vector_type(4)))  float  v4f;

#define LDSS 40                 // bf16 elems per LDS row (32 data + 8 pad -> 80B stride)
#define BUFE (128 * LDSS)       // elements per tile buffer
#define BUFB (BUFE * 2)         // bytes per tile buffer

// ---- CDNA5 async copy: 16B global -> LDS on the ASYNC pipe (ASYNCcnt) ----
__device__ __forceinline__ void async_cp16(uint32_t lds_off, uint64_t gaddr) {
    asm volatile("global_load_async_to_lds_b128 %0, %1, off"
                 :: "v"(lds_off), "v"(gaddr) : "memory");
}
__device__ __forceinline__ void wait_async_le4() {
    asm volatile("s_wait_asynccnt 0x4" ::: "memory");
}
__device__ __forceinline__ void wait_async_0() {
    asm volatile("s_wait_asynccnt 0x0" ::: "memory");
}

// NT GEMM on bf16 operands: acc[row,col] = sum_k Ag[row,k]*Bg[col,k]  (f32 accum WMMA)
// MODE 0: Cf = acc                (h = y16 @ W0_16^T)
// MODE 1: Ch = bf16(acc - aux)    (resid16 = x16 @ A16^T - y)
// MODE 2: Cf = aux - acc          (V = x - resid16 @ Wt16^T)
template <int MODE>
__global__ __launch_bounds__(256)
void gemm_nt_bf16(float* __restrict__ Cf, __bf16* __restrict__ Ch,
                  const __bf16* __restrict__ Ag, const __bf16* __restrict__ Bg,
                  const float* __restrict__ aux, int K, int ldc)
{
    __shared__ __align__(16) __bf16 sA[2 * BUFE];
    __shared__ __align__(16) __bf16 sB[2 * BUFE];

    const int tid   = threadIdx.x;
    const int lane  = tid & 31;
    const int wave  = tid >> 5;
    const int waveM = wave & 3;      // 4 row-groups of 32 rows
    const int waveN = wave >> 2;     // 2 col-groups of 64 cols
    const int rowBase = blockIdx.y << 7;
    const int colBase = blockIdx.x << 7;

    // async tile copy: 2 threads per row, each moves 32B (2 x b128) per operand
    const int lrow = tid >> 1;           // 0..127
    const int lsub = (tid & 1) << 4;     // element offset 0 or 16 within 32-K slab
    const uint64_t gA = (uint64_t)(uintptr_t)(Ag + (size_t)(rowBase + lrow) * (size_t)K + lsub);
    const uint64_t gB = (uint64_t)(uintptr_t)(Bg + (size_t)(colBase + lrow) * (size_t)K + lsub);
    const uint32_t lA = (uint32_t)(uintptr_t)&sA[lrow * LDSS + lsub];
    const uint32_t lB = (uint32_t)(uintptr_t)&sB[lrow * LDSS + lsub];

    v8f acc[2][4];
#pragma unroll
    for (int m = 0; m < 2; ++m)
#pragma unroll
        for (int n = 0; n < 4; ++n)
#pragma unroll
            for (int i = 0; i < 8; ++i) acc[m][n][i] = 0.0f;

    const int half8 = (lane >> 4) << 3;  // 0 or 8 (lane-group K select)
    const int l16   = lane & 15;
    const __bf16* pa0 = &sA[(waveM * 32 + l16) * LDSS];
    const __bf16* pb0 = &sB[(waveN * 64 + l16) * LDSS];

    const int nsteps = K >> 5;

    // prologue: fill buffer 0
    async_cp16(lA, gA);
    async_cp16(lA + 16, gA + 16);
    async_cp16(lB, gB);
    async_cp16(lB + 16, gB + 16);

    for (int s = 0; s < nsteps; ++s) {
        const int cur = s & 1;
        if (s + 1 < nsteps) {
            // prefetch next K-slab into the other buffer (that buffer's readers
            // finished before the trailing barrier of the previous iteration)
            const uint64_t gk = (uint64_t)((s + 1) << 6);   // 32 elems * 2B
            const uint32_t lb = (uint32_t)((cur ^ 1) * BUFB);
            async_cp16(lA + lb, gA + gk);
            async_cp16(lA + lb + 16, gA + gk + 16);
            async_cp16(lB + lb, gB + gk);
            async_cp16(lB + lb + 16, gB + gk + 16);
            wait_async_le4();          // all but the 4 just-issued are complete
        } else {
            wait_async_0();
        }
        __syncthreads();               // cur buffer visible to all waves

        const __bf16* pa = pa0 + cur * BUFE;
        const __bf16* pb = pb0 + cur * BUFE;

        // A fragment (ISA 16-bit A 16x32): lane<16 -> K 0..7 & 16..23, lane>=16 -> K 8..15 & 24..31
        v16bf af[2];
#pragma unroll
        for (int mt = 0; mt < 2; ++mt) {
            const __bf16* p = pa + mt * 16 * LDSS;
            v8bf lo = *(const v8bf*)(p + half8);
            v8bf hi = *(const v8bf*)(p + 16 + half8);
            af[mt] = __builtin_shufflevector(lo, hi, 0,1,2,3,4,5,6,7,8,9,10,11,12,13,14,15);
        }
        // B fragment: same K order as A: lane<16 -> K 0..7 & 16..23, lane>=16 -> K 8..15 & 24..31
        v16bf bfr[4];
#pragma unroll
        for (int nt = 0; nt < 4; ++nt) {
            const __bf16* p = pb + nt * 16 * LDSS;
            v8bf lo = *(const v8bf*)(p + half8);
            v8bf hi = *(const v8bf*)(p + 16 + half8);
            bfr[nt] = __builtin_shufflevector(lo, hi, 0,1,2,3,4,5,6,7,8,9,10,11,12,13,14,15);
        }
#pragma unroll
        for (int mt = 0; mt < 2; ++mt)
#pragma unroll
            for (int nt = 0; nt < 4; ++nt)
                acc[mt][nt] = __builtin_amdgcn_wmma_f32_16x16x32_bf16(
                    false, af[mt], false, bfr[nt], (short)0, acc[mt][nt], false, false);

        __syncthreads();               // readers done before buffer is overwritten
    }

    // epilogue: C/D layout — VGPR i: M = i + (lane>=16 ? 8 : 0), N = lane & 15
    const int rloc = (lane >> 4) << 3;
#pragma unroll
    for (int mt = 0; mt < 2; ++mt)
#pragma unroll
        for (int nt = 0; nt < 4; ++nt)
#pragma unroll
            for (int i = 0; i < 8; ++i) {
                int row = rowBase + waveM * 32 + mt * 16 + rloc + i;
                int col = colBase + waveN * 64 + nt * 16 + l16;
                size_t off = (size_t)row * (size_t)ldc + (size_t)col;
                float r = acc[mt][nt][i];
                if (MODE == 0) {
                    Cf[off] = r;
                } else if (MODE == 1) {
                    Ch[off] = (__bf16)(r - aux[off]);
                } else {
                    Cf[off] = aux[off] - r;
                }
            }
}

// fp32 -> bf16 bulk convert (once per launch for weights / y; 4 elems per thread)
__global__ __launch_bounds__(256)
void f32_to_bf16(__bf16* __restrict__ out, const float* __restrict__ in)
{
    size_t i = ((size_t)blockIdx.x * 256 + threadIdx.x) * 4;
    v4f a = *(const v4f*)(in + i);
    v4bf r;
    r[0] = (__bf16)a.x; r[1] = (__bf16)a.y; r[2] = (__bf16)a.z; r[3] = (__bf16)a.w;
    *(v4bf*)(out + i) = r;
}

// Per-row CPSS shrink: exact k-th largest |v| via 4-pass radix select on float bits,
// then x = (|v| >= thr) ? v : sign(v)*max(|v|-beta, 0). Emits fp32 x and bf16 x copy.
__global__ __launch_bounds__(256)
void cpss_shrink(float* __restrict__ x, __bf16* __restrict__ x16,
                 const float* __restrict__ V, const float* __restrict__ betas,
                 int bidx, int k)
{
    __shared__ unsigned hist[256];
    __shared__ unsigned s_pref;
    __shared__ int s_k;

    const int tid = threadIdx.x;
    const size_t base = (size_t)blockIdx.x * 2048 + tid;

    float v[8]; unsigned u[8];
#pragma unroll
    for (int j = 0; j < 8; ++j) {
        v[j] = V[base + (size_t)j * 256];
        u[j] = __float_as_uint(fabsf(v[j]));   // non-negative float: bit order == value order
    }

    unsigned prefix = 0; int kk = k;
    for (int shift = 24; shift >= 0; shift -= 8) {
        hist[tid] = 0;
        __syncthreads();
        unsigned mask = (shift == 24) ? 0u : (0xFFFFFFFFu << (shift + 8));
#pragma unroll
        for (int j = 0; j < 8; ++j)
            if ((u[j] & mask) == prefix)
                atomicAdd(&hist[(u[j] >> shift) & 255u], 1u);
        __syncthreads();
        if (tid == 0) {
            int cum = 0; int b = 255;
            for (; b > 0; --b) { int c = (int)hist[b]; if (cum + c >= kk) break; cum += c; }
            s_pref = prefix | ((unsigned)b << shift);
            s_k = kk - cum;
        }
        __syncthreads();
        prefix = s_pref; kk = s_k;
        __syncthreads();
    }

    const float thr  = __uint_as_float(prefix);
    const float beta = betas[bidx];
#pragma unroll
    for (int j = 0; j < 8; ++j) {
        float val  = v[j];
        float a    = fabsf(val);
        float soft = copysignf(fmaxf(a - beta, 0.0f), val);
        float o    = (a >= thr) ? val : soft;
        x[base + (size_t)j * 256]   = o;
        x16[base + (size_t)j * 256] = (__bf16)o;
    }
}

extern "C" void kernel_launch(void* const* d_in, const int* in_sizes, int n_in,
                              void* d_out, int out_size, void* d_ws, size_t ws_size,
                              hipStream_t stream)
{
    const float* y     = (const float*)d_in[0];   // [8192, 1024]
    const float* A     = (const float*)d_in[1];   // [1024, 2048]
    const float* Ws    = (const float*)d_in[2];   // [6, 2048, 1024]
    const float* betas = (const float*)d_in[3];   // [6]

    float* x = (float*)d_out;                     // [8192, 2048] iterated state (fp32)

    // scratch layout
    char* ws = (char*)d_ws;
    float*  V       = (float*)ws;                         ws += (size_t)8192 * 2048 * 4;  // 64 MB
    __bf16* x16     = (__bf16*)ws;                        ws += (size_t)8192 * 2048 * 2;  // 32 MB
    __bf16* resid16 = (__bf16*)ws;                        ws += (size_t)8192 * 1024 * 2;  // 16 MB
    __bf16* y16     = (__bf16*)ws;                        ws += (size_t)8192 * 1024 * 2;  // 16 MB
    __bf16* A16     = (__bf16*)ws;                        ws += (size_t)1024 * 2048 * 2;  //  4 MB
    __bf16* Ws16    = (__bf16*)ws;                                                        // 24 MB

    const dim3 blk(256);
    const dim3 gN(16, 64);   // 2048-col output
    const dim3 gM(8, 64);    // 1024-col output
    // k_t = int(min(0.012*t, 0.12) * 2048)
    static const int kidx[6] = {24, 24, 49, 73, 98, 122};

    // one-time (per launch) bf16 conversions — same rounding as converting inside the GEMM
    f32_to_bf16<<<dim3(8192 * 1024 / 1024), blk, 0, stream>>>(y16, y);
    f32_to_bf16<<<dim3(1024 * 2048 / 1024), blk, 0, stream>>>(A16, A);
    f32_to_bf16<<<dim3(6 * 2048 * 1024 / 1024), blk, 0, stream>>>(Ws16, Ws);

    // x = shrink(y @ W0^T, betas[0], t=1)
    gemm_nt_bf16<0><<<gN, blk, 0, stream>>>(V, nullptr, y16, Ws16, nullptr, 1024, 2048);
    cpss_shrink<<<dim3(8192), blk, 0, stream>>>(x, x16, V, betas, 0, kidx[0]);

    for (int t = 1; t <= 5; ++t) {
        // resid16 = bf16(x @ A^T - y)
        gemm_nt_bf16<1><<<gM, blk, 0, stream>>>(nullptr, resid16, x16, A16, y, 2048, 1024);
        // V = x - resid @ Wt^T
        gemm_nt_bf16<2><<<gN, blk, 0, stream>>>(V, nullptr, resid16,
                                                Ws16 + (size_t)t * 2048 * 1024, x, 1024, 2048);
        // x = shrink(V, betas[t], t)
        cpss_shrink<<<dim3(8192), blk, 0, stream>>>(x, x16, V, betas, t, kidx[t]);
    }
}
